// Self_Attention_24524263260928
// MI455X (gfx1250) — matrix operand order repaired and checked by reading, not compiled, not run on hardware
//
#include <hip/hip_runtime.h>

#define B_  4
#define C_  256
#define C8_ 32
#define N_  4096
#define PROW 40   // padded LDS row stride (ushorts); 80B keeps 16B alignment for b128 LDS ops
#define LOG2E 1.44269504088896340736f

typedef __attribute__((ext_vector_type(16))) __bf16 v16bf;
typedef __attribute__((ext_vector_type(8)))  float  v8f;

union Frag {
  v16bf v;
  unsigned short s[16];
  uint4 q[2];
};
union Pack8 {
  unsigned short s[8];
  uint4 q;
};

__device__ __forceinline__ unsigned short f2bf(float f) {
  return __builtin_bit_cast(unsigned short, (__bf16)f);   // native v_cvt, RNE
}

#define WMMA_BF16(a, b, c) \
  __builtin_amdgcn_wmma_f32_16x16x32_bf16(false, (a), false, (b), (short)0, (c), false, false)

// ---------------------------------------------------------------------------
// Kernel 0: one-time weight conversion to bf16.
// ---------------------------------------------------------------------------
__global__ __launch_bounds__(256) void k_prep(
    const float* __restrict__ Wf, const float* __restrict__ Wg,
    const float* __restrict__ Wh, const float* __restrict__ Wv,
    unsigned short* __restrict__ WfB, unsigned short* __restrict__ WgB,
    unsigned short* __restrict__ WhB, unsigned short* __restrict__ WvB) {
  const int t = blockIdx.x * 256 + threadIdx.x;
  WfB[t] = f2bf(Wf[t]);
  WgB[t] = f2bf(Wg[t]);
  WhB[t] = f2bf(Wh[t]);
  WvB[t] = f2bf(Wv[t]);
}

// ---------------------------------------------------------------------------
// Kernel 1: projections.  f stored transposed AND pre-scaled by log2(e):
// fT[b][n][c] (bf16).  g transposed gT[b][n][c], h row-major hM[b][c][n].
// One wave per (batch, 16-pixel tile): [32,256] x [256,16], 8 K-chunks.
// ---------------------------------------------------------------------------
__global__ __launch_bounds__(128) void k_proj(
    const float* __restrict__ x,
    const unsigned short* __restrict__ WfB, const float* __restrict__ bf,
    const unsigned short* __restrict__ WgB, const float* __restrict__ bg,
    const unsigned short* __restrict__ WhB, const float* __restrict__ bh,
    unsigned short* __restrict__ fT, unsigned short* __restrict__ gT,
    unsigned short* __restrict__ hM) {
  const int wave = threadIdx.x >> 5;
  const int lane = threadIdx.x & 31;
  const int n    = lane & 15;        // B-fragment column / A-fragment row
  const int half = lane >> 4;
  const int w    = blockIdx.x * 4 + wave;
  const int b    = w >> 8;
  const int n0   = (w & 255) << 4;

  v8f af0 = {}, af1 = {}, ag0 = {}, ag1 = {}, ah0 = {}, ah1 = {};

  for (int kc = 0; kc < 8; ++kc) {
    // B fragment: element e -> channel c = kc*32 + half*16 + e, pixel n0+n
    Frag bx;
    {
      const float* xp = x + ((size_t)b * C_ + kc * 32 + half * 16) * N_ + n0 + n;
      #pragma unroll
      for (int e = 0; e < 16; ++e) bx.s[e] = f2bf(xp[(size_t)e * N_]);
    }
    // A fragment (bf16 weights): element e -> k = kc*32+half*16+e
    auto loadW = [&](const unsigned short* W, int mt) {
      Frag a;
      const unsigned short* p = W + (mt * 16 + n) * C_ + kc * 32 + half * 16;
      a.q[0] = *reinterpret_cast<const uint4*>(p);
      a.q[1] = *reinterpret_cast<const uint4*>(p + 8);
      return a;
    };
    af0 = WMMA_BF16(loadW(WfB, 0).v, bx.v, af0);
    af1 = WMMA_BF16(loadW(WfB, 1).v, bx.v, af1);
    ag0 = WMMA_BF16(loadW(WgB, 0).v, bx.v, ag0);
    ag1 = WMMA_BF16(loadW(WgB, 1).v, bx.v, ag1);
    ah0 = WMMA_BF16(loadW(WhB, 0).v, bx.v, ah0);
    ah1 = WMMA_BF16(loadW(WhB, 1).v, bx.v, ah1);
  }

  // C/D layout: VGPR r holds row  c = mt*16 + 8*half + r  for column n0+n.
  const size_t row = ((size_t)b * N_ + n0 + n) * C8_;
  auto storeT = [&](unsigned short* T, const v8f& a0, const v8f& a1,
                    const float* bias, float scale) {
    Pack8 p;
    #pragma unroll
    for (int r = 0; r < 8; ++r) p.s[r] = f2bf((a0[r] + bias[8 * half + r]) * scale);
    *reinterpret_cast<uint4*>(T + row + 8 * half) = p.q;
    #pragma unroll
    for (int r = 0; r < 8; ++r) p.s[r] = f2bf((a1[r] + bias[16 + 8 * half + r]) * scale);
    *reinterpret_cast<uint4*>(T + row + 16 + 8 * half) = p.q;
  };
  storeT(fT, af0, af1, bf, LOG2E);   // pre-scaled: S' = log2(e) * S
  storeT(gT, ag0, ag1, bg, 1.0f);
  #pragma unroll
  for (int r = 0; r < 8; ++r) {
    const int c0 = 8 * half + r;
    hM[((size_t)b * C8_ + c0) * N_ + n0 + n] = f2bf(ah0[r] + bh[c0]);
    const int c1 = 16 + 8 * half + r;
    hM[((size_t)b * C8_ + c1) * N_ + n0 + n] = f2bf(ah1[r] + bh[c1]);
  }
}

// ---------------------------------------------------------------------------
// Kernel 2 (pass A): per-row stats in base-2:  q[k] = m + log2(sum_i 2^(S'-m)).
// One wave per 16-row k-tile; A(fT rows) loaded once, sweep all i with WMMA.
// ---------------------------------------------------------------------------
__global__ __launch_bounds__(128) void k_stats(
    const unsigned short* __restrict__ fT,
    const unsigned short* __restrict__ gT,
    float* __restrict__ q) {
  const int wave = threadIdx.x >> 5;
  const int lane = threadIdx.x & 31;
  const int n    = lane & 15;
  const int half = lane >> 4;
  const int w    = blockIdx.x * 4 + wave;
  const int b    = w >> 8;
  const int k0   = (w & 255) << 4;

  Frag afr;
  {
    const unsigned short* p = fT + ((size_t)b * N_ + k0 + n) * C8_;
    afr.q[0] = *reinterpret_cast<const uint4*>(p + half * 16);
    afr.q[1] = *reinterpret_cast<const uint4*>(p + half * 16 + 8);
  }

  float mx[8], sm[8];
  #pragma unroll
  for (int r = 0; r < 8; ++r) { mx[r] = -3.0e38f; sm[r] = 0.0f; }

  const v8f vz = {};
  const unsigned short* gbase = gT + (size_t)b * N_ * C8_ + half * 16;
  #pragma unroll 2
  for (int i0 = 0; i0 < N_; i0 += 16) {
    Frag bg;
    const unsigned short* p = gbase + (size_t)(i0 + n) * C8_;
    bg.q[0] = *reinterpret_cast<const uint4*>(p);
    bg.q[1] = *reinterpret_cast<const uint4*>(p + 8);
    v8f s = WMMA_BF16(afr.v, bg.v, vz);
    #pragma unroll
    for (int r = 0; r < 8; ++r) {   // online max/sum; row = k0 + 8*half + r
      const float sv = s[r];
      const float mo = mx[r];
      const float mn = fmaxf(mo, sv);
      sm[r] = sm[r] * exp2f(mo - mn) + exp2f(sv - mn);
      mx[r] = mn;
    }
  }
  // reduce across the 16 lanes of each half (columns of the row)
  #pragma unroll
  for (int d = 1; d < 16; d <<= 1) {
    #pragma unroll
    for (int r = 0; r < 8; ++r) {
      const float mo = __shfl_xor(mx[r], d, 32);
      const float so = __shfl_xor(sm[r], d, 32);
      const float mn = fmaxf(mx[r], mo);
      sm[r] = sm[r] * exp2f(mx[r] - mn) + so * exp2f(mo - mn);
      mx[r] = mn;
    }
  }
  if (n == 0) {
    #pragma unroll
    for (int r = 0; r < 8; ++r)
      q[(size_t)b * N_ + k0 + 8 * half + r] = mx[r] + log2f(sm[r]);
  }
}

// ---------------------------------------------------------------------------
// Kernel 3 (pass B): fully fused  out = gamma * (Wv @ (h @ 2^(S' - q)) + bv).
// One wave per 16-col i-tile; k-loop unrolled 2x with double-buffered LDS
// staging tiles so chunk B's score WMMA + exp work overlaps chunk A's
// LDS round-trip and accumulate WMMAs.  Epilogue: 16 Wv WMMAs (K=C8=32)
// with the bias as the C operand, output written straight from D fragments.
// ---------------------------------------------------------------------------
__global__ __launch_bounds__(128) void k_attn(
    const unsigned short* __restrict__ fT,
    const unsigned short* __restrict__ gT,
    const unsigned short* __restrict__ hM,
    const float* __restrict__ q,
    const unsigned short* __restrict__ WvB, const float* __restrict__ bv,
    const float* __restrict__ gamma,
    float* __restrict__ out) {
  __shared__ float qs[N_];
  __shared__ __align__(16) unsigned short pb[4][2][16][PROW];

  const int wave = threadIdx.x >> 5;
  const int lane = threadIdx.x & 31;
  const int n    = lane & 15;
  const int half = lane >> 4;
  const int b    = blockIdx.x >> 6;
  const int i0   = (((blockIdx.x & 63) * 4 + wave)) << 4;

  for (int t = threadIdx.x; t < N_; t += 128) qs[t] = q[(size_t)b * N_ + t];
  __syncthreads();

  Frag bg;   // g columns for this i-tile: fixed for the whole k sweep
  {
    const unsigned short* p = gT + ((size_t)b * N_ + i0 + n) * C8_ + half * 16;
    bg.q[0] = *reinterpret_cast<const uint4*>(p);
    bg.q[1] = *reinterpret_cast<const uint4*>(p + 8);
  }

  unsigned short* prow0 = &pb[wave][0][n][0];
  unsigned short* prow1 = &pb[wave][1][n][0];
  const v8f vz = {};
  v8f acc0 = {}, acc1 = {};
  const unsigned short* fbase = fT + (size_t)b * N_ * C8_;
  const unsigned short* hbase = hM + (size_t)b * C8_ * N_;

  auto chunk = [&](int k0, unsigned short* prow) {
    #pragma unroll
    for (int t = 0; t < 2; ++t) {
      const int kk = k0 + 16 * t;
      Frag afr;
      const unsigned short* p = fbase + (size_t)(kk + n) * C8_;
      afr.q[0] = *reinterpret_cast<const uint4*>(p + half * 16);
      afr.q[1] = *reinterpret_cast<const uint4*>(p + half * 16 + 8);
      v8f s = WMMA_BF16(afr.v, bg.v, vz);      // S'[kk+.., i0+..]
      const float4* qp = reinterpret_cast<const float4*>(&qs[kk + 8 * half]);
      const float4 q0 = qp[0], q1 = qp[1];
      float pv[8];
      pv[0] = exp2f(s[0] - q0.x); pv[1] = exp2f(s[1] - q0.y);
      pv[2] = exp2f(s[2] - q0.z); pv[3] = exp2f(s[3] - q0.w);
      pv[4] = exp2f(s[4] - q1.x); pv[5] = exp2f(s[5] - q1.y);
      pv[6] = exp2f(s[6] - q1.z); pv[7] = exp2f(s[7] - q1.w);
      unsigned short* dst = prow + 8 * half + 16 * t;
      #pragma unroll
      for (int j = 0; j < 4; ++j)
        reinterpret_cast<unsigned int*>(dst)[j] =
            (unsigned int)f2bf(pv[2 * j]) | ((unsigned int)f2bf(pv[2 * j + 1]) << 16);
    }
    Frag bp;   // P B-fragment: 32 k-rows x 16 cols (LDS relayout, in-order DS)
    bp.q[0] = *reinterpret_cast<const uint4*>(prow + half * 8);
    bp.q[1] = *reinterpret_cast<const uint4*>(prow + 16 + half * 8);
    Frag ha0, ha1;
    {
      const unsigned short* p = hbase + (size_t)n * N_ + k0;
      ha0.q[0] = *reinterpret_cast<const uint4*>(p + half * 8);
      ha0.q[1] = *reinterpret_cast<const uint4*>(p + 16 + half * 8);
      const unsigned short* p2 = p + (size_t)16 * N_;
      ha1.q[0] = *reinterpret_cast<const uint4*>(p2 + half * 8);
      ha1.q[1] = *reinterpret_cast<const uint4*>(p2 + 16 + half * 8);
    }
    acc0 = WMMA_BF16(ha0.v, bp.v, acc0);   // sa rows 0..15
    acc1 = WMMA_BF16(ha1.v, bp.v, acc1);   // sa rows 16..31
  };

  for (int k0 = 0; k0 < N_; k0 += 64) {
    chunk(k0,      prow0);
    chunk(k0 + 32, prow1);
  }

  // relay sa tile (32 x 16) through LDS into a B fragment (K = C8 = 32)
  {
    unsigned short* d0 = prow0 + 8 * half;
    #pragma unroll
    for (int j = 0; j < 4; ++j)
      reinterpret_cast<unsigned int*>(d0)[j] =
          (unsigned int)f2bf(acc0[2 * j]) | ((unsigned int)f2bf(acc0[2 * j + 1]) << 16);
    unsigned short* d1 = prow0 + 16 + 8 * half;
    #pragma unroll
    for (int j = 0; j < 4; ++j)
      reinterpret_cast<unsigned int*>(d1)[j] =
          (unsigned int)f2bf(acc1[2 * j]) | ((unsigned int)f2bf(acc1[2 * j + 1]) << 16);
  }
  Frag bsa;
  bsa.q[0] = *reinterpret_cast<const uint4*>(prow0 + half * 8);
  bsa.q[1] = *reinterpret_cast<const uint4*>(prow0 + 16 + half * 8);

  const float gm = gamma[0];
  for (int mt = 0; mt < 16; ++mt) {     // Wv projection, bias via C operand
    const int co0 = mt * 16;
    Frag awv;
    const unsigned short* wp = WvB + (co0 + n) * C8_ + half * 8;
    awv.q[0] = *reinterpret_cast<const uint4*>(wp);
    awv.q[1] = *reinterpret_cast<const uint4*>(wp + 16);
    v8f cb;
    #pragma unroll
    for (int r = 0; r < 8; ++r) cb[r] = bv[co0 + 8 * half + r];
    v8f d = WMMA_BF16(awv.v, bsa.v, cb);
    #pragma unroll
    for (int r = 0; r < 8; ++r)
      out[((size_t)b * C_ + co0 + 8 * half + r) * N_ + i0 + n] = gm * d[r];
  }
}

// ---------------------------------------------------------------------------
extern "C" void kernel_launch(void* const* d_in, const int* in_sizes, int n_in,
                              void* d_out, int out_size, void* d_ws, size_t ws_size,
                              hipStream_t stream) {
  (void)in_sizes; (void)n_in; (void)out_size; (void)ws_size;
  const float* x     = (const float*)d_in[0];
  const float* Wf    = (const float*)d_in[1];
  const float* bf    = (const float*)d_in[2];
  const float* Wg    = (const float*)d_in[3];
  const float* bg    = (const float*)d_in[4];
  const float* Wh    = (const float*)d_in[5];
  const float* bh    = (const float*)d_in[6];
  const float* Wv    = (const float*)d_in[7];
  const float* bv    = (const float*)d_in[8];
  const float* gamma = (const float*)d_in[9];

  char* ws = (char*)d_ws;
  unsigned short* fT  = (unsigned short*)(ws);                        // 1 MB  [B][N][32] bf16 (log2e-scaled)
  unsigned short* gT  = (unsigned short*)(ws + (1u << 20));           // 1 MB  [B][N][32] bf16
  unsigned short* hM  = (unsigned short*)(ws + (2u << 20));           // 1 MB  [B][32][N] bf16
  float*          q   = (float*)        (ws + (3u << 20));            // 64 KB [B][N] f32
  unsigned short* WfB = (unsigned short*)(ws + (3u << 20) + 65536);   // 16 KB each
  unsigned short* WgB = WfB + C8_ * C_;
  unsigned short* WhB = WgB + C8_ * C_;
  unsigned short* WvB = WhB + C8_ * C_;

  k_prep <<<32, 256, 0, stream>>>(Wf, Wg, Wh, Wv, WfB, WgB, WhB, WvB);
  k_proj <<<256, 128, 0, stream>>>(x, WfB, bf, WgB, bg, WhB, bh, fT, gT, hM);
  k_stats<<<256, 128, 0, stream>>>(fT, gT, q);
  k_attn <<<256, 128, 0, stream>>>(fT, gT, hM, q, WvB, bv, gamma, (float*)d_out);
}
